// VQVAE_12910671692140
// MI455X (gfx1250) — matrix operand order repaired and checked by reading, not compiled, not run on hardware
//
#include <hip/hip_runtime.h>
#include <stdint.h>

// ---------------------------------------------------------------------------
// VQ-VAE forward for MI455X (gfx1250, wave32, WMMA).
// All dense layers run through v_wmma_f32_16x16x32_bf16 (bf16 in, f32 accum).
// VQ argmin is fused into the distance GEMM epilogue (no 536MB dist matrix).
// GEMM/VQ tiles are streamed with GLOBAL_LOAD_ASYNC_TO_LDS_B128 (inline asm,
// ASYNCcnt producer/consumer with double buffering) on gfx1250.
// ---------------------------------------------------------------------------

typedef __attribute__((ext_vector_type(16))) __bf16 v16bf;
typedef __attribute__((ext_vector_type(8)))  float  v8f;
typedef unsigned short u16t;

#define BDIM   4096
#define CONDD  256
#define ED     128
#define LD     16
#define KCODE  2048
#define PD     256
#define LE     2048      // L*E
#define NROWS  65536     // B*L

#if defined(__gfx1250__)
#define USE_ASYNC_LDS 1
#else
#define USE_ASYNC_LDS 0
#endif

union FragBF { uint4 q[2]; v16bf v; };
union AccF   { v8f v; float f[8]; };

__device__ __forceinline__ u16t f2bf(float f) {
  union { float f; unsigned u; } c; c.f = f;
  unsigned r = c.u + 0x7FFFu + ((c.u >> 16) & 1u);   // RNE
  return (u16t)(r >> 16);
}

#if USE_ASYNC_LDS
// Async DMA: 16B per lane, global -> LDS, tracked by ASYNCcnt.
// VDST vgpr = wave-relative LDS byte address (flat LDS addr truncates to [31:0]),
// VADDR vgpr pair = 64-bit global address, 'off' = no SADDR.
__device__ __forceinline__ void async_g2l_b128(const void* gsrc, void* ldst) {
  asm volatile("global_load_async_to_lds_b128 %0, %1, off"
               :
               : "v"((unsigned)(size_t)ldst),
                 "v"((unsigned long long)(size_t)gsrc)
               : "memory");
}
__device__ __forceinline__ void wait_async0() {
#if __has_builtin(__builtin_amdgcn_s_wait_asynccnt)
  __builtin_amdgcn_s_wait_asynccnt(0);
#else
  asm volatile("s_wait_asynccnt 0x0" ::: "memory");
#endif
}
#endif

// -------------------- f32 -> bf16 elementwise --------------------
__global__ void cvt_f32_bf16(const float* __restrict__ in, u16t* __restrict__ out, int n) {
  int i = blockIdx.x * blockDim.x + threadIdx.x;
  if (i < n) out[i] = f2bf(in[i]);
}

// -------------------- weight transpose+convert: W[K][N] -> WT[N][K] bf16 ----
__global__ void wxpose_bf16(const float* __restrict__ in, u16t* __restrict__ out,
                            int Kd, int N) {
  __shared__ float tile[32][33];
  const int kt = blockIdx.y * 32, nt = blockIdx.x * 32;
  const int tx = threadIdx.x, ty = threadIdx.y;            // 32 x 8
  #pragma unroll
  for (int r = ty; r < 32; r += 8)
    tile[r][tx] = in[(size_t)(kt + r) * N + nt + tx];
  __syncthreads();
  #pragma unroll
  for (int r = ty; r < 32; r += 8)
    out[(size_t)(nt + r) * Kd + kt + tx] = f2bf(tile[tx][r]);
}

// -------------------- codebook: bf16 copy + squared norms --------------------
__global__ void prep_codebook(const float* __restrict__ cb, u16t* __restrict__ cb_bf,
                              float* __restrict__ cnorm) {
  const int c = blockIdx.x, t = threadIdx.x;               // 128 threads
  __shared__ float s[128];
  float v = cb[(size_t)c * ED + t];
  cb_bf[(size_t)c * ED + t] = f2bf(v);
  s[t] = v * v;
  __syncthreads();
  for (int o = 64; o > 0; o >>= 1) { if (t < o) s[t] += s[t + o]; __syncthreads(); }
  if (t == 0) cnorm[c] = s[0];
}

// -------------------- zero fill (float4) --------------------
__global__ void zero_f32v4(float4* __restrict__ p, size_t n4) {
  size_t i = (size_t)blockIdx.x * blockDim.x + threadIdx.x;
  size_t stride = (size_t)gridDim.x * blockDim.x;
  float4 z; z.x = z.y = z.z = z.w = 0.f;
  for (; i < n4; i += stride) p[i] = z;
}

// ---------------------------------------------------------------------------
// Generic bf16 WMMA GEMM with fused bias + leaky_relu(0.2).
//   X  : [M, Kd] bf16 row-major
//   WT : [N, Kd] bf16 (weight transposed)
//   out = leaky_relu(X @ W + bias) -> bf16 always, f32 optionally
// BM=BN=128, BK=32, 256 threads = 8 waves (2 M-waves x 4 N-waves),
// each wave owns a 64x32 region = 4x2 WMMA 16x16 tiles.
// Tiles stream via async-to-LDS with double buffering on gfx1250.
// ---------------------------------------------------------------------------
__global__ __launch_bounds__(256) void gemm_bf16_lrelu(
    const u16t* __restrict__ X, const u16t* __restrict__ WT,
    const float* __restrict__ bias,
    u16t* __restrict__ outBF, float* __restrict__ outF32,
    int N, int Kd)
{
  __shared__ u16t As[2][128 * 40];   // [row][k], pad 40 to dodge bank conflicts
  __shared__ u16t Bs[2][128 * 40];   // [col][k]
  const int t    = threadIdx.x;
  const int lane = t & 31;
  const int wave = t >> 5;
  const int wm   = wave & 1;          // 0..1  (M)
  const int wn   = wave >> 1;         // 0..3  (N)
  const int l16  = lane & 15;
  const int hl   = lane >> 4;
  const size_t rowBlock = (size_t)blockIdx.y * 128;
  const size_t colBlock = (size_t)blockIdx.x * 128;

  const v8f vzero = {0.f,0.f,0.f,0.f,0.f,0.f,0.f,0.f};
  AccF acc[4][2];
  #pragma unroll
  for (int m = 0; m < 4; ++m)
    #pragma unroll
    for (int n = 0; n < 2; ++n) acc[m][n].v = vzero;

#if USE_ASYNC_LDS
  auto issueT = [&](int p, int kb) {
    #pragma unroll
    for (int i = 0; i < 2; ++i) {
      int f = t + i * 256;            // 0..511 -> 128 rows x 4 k-segments
      int r = f >> 2, ks = f & 3;
      async_g2l_b128(X  + (rowBlock + r) * (size_t)Kd + kb + ks * 8,
                     &As[p][r * 40 + ks * 8]);
      async_g2l_b128(WT + (colBlock + r) * (size_t)Kd + kb + ks * 8,
                     &Bs[p][r * 40 + ks * 8]);
    }
  };
  issueT(0, 0);
  wait_async0();
  __syncthreads();
  int p = 0;
  for (int kb = 0; kb < Kd; kb += 32, p ^= 1) {
    if (kb + 32 < Kd) issueT(p ^ 1, kb + 32);   // prefetch next tile
#else
  int p = 0;
  for (int kb = 0; kb < Kd; kb += 32) {
    __syncthreads();
    #pragma unroll
    for (int i = 0; i < 2; ++i) {
      int f = t + i * 256;
      int r = f >> 2, ks = f & 3;
      *(uint4*)&As[0][r * 40 + ks * 8] =
          *(const uint4*)(X  + (rowBlock + r) * (size_t)Kd + kb + ks * 8);
      *(uint4*)&Bs[0][r * 40 + ks * 8] =
          *(const uint4*)(WT + (colBlock + r) * (size_t)Kd + kb + ks * 8);
    }
    __syncthreads();
#endif

    FragBF a[4], b[2];
    #pragma unroll
    for (int m = 0; m < 4; ++m) {
      // A 16x32: lane<16 holds K{hl*8..+7, 16+hl*8..+7} of row (l16)
      const u16t* ap = &As[p][(wm * 64 + m * 16 + l16) * 40 + hl * 8];
      a[m].q[0] = *(const uint4*)ap;
      a[m].q[1] = *(const uint4*)(ap + 16);
    }
    #pragma unroll
    for (int n = 0; n < 2; ++n) {
      // B 32x16: lane holds K{hl*8..+7, 16+hl*8..+7} of column (l16)
      const u16t* bp = &Bs[p][(wn * 32 + n * 16 + l16) * 40 + hl * 8];
      b[n].q[0] = *(const uint4*)bp;
      b[n].q[1] = *(const uint4*)(bp + 16);
    }
    #pragma unroll
    for (int m = 0; m < 4; ++m)
      #pragma unroll
      for (int n = 0; n < 2; ++n)
        acc[m][n].v = __builtin_amdgcn_wmma_f32_16x16x32_bf16(
            false, a[m].v, false, b[n].v, (short)0, acc[m][n].v, false, false);

#if USE_ASYNC_LDS
    wait_async0();
    __syncthreads();
#endif
  }

  // epilogue: D elem v -> (M = v + 8*hl, N = l16)
  #pragma unroll
  for (int m = 0; m < 4; ++m) {
    const size_t row = rowBlock + wm * 64 + m * 16 + hl * 8;
    #pragma unroll
    for (int n = 0; n < 2; ++n) {
      const size_t col = colBlock + wn * 32 + n * 16 + l16;
      const float bv = bias[col];
      #pragma unroll
      for (int v = 0; v < 8; ++v) {
        float x = acc[m][n].f[v] + bv;
        x = (x > 0.f) ? x : 0.2f * x;
        size_t o = (row + v) * (size_t)N + col;
        outBF[o] = f2bf(x);
        if (outF32) outF32[o] = x;
      }
    }
  }
}

// ---------------------------------------------------------------------------
// Fused VQ: dist(r,c) = ||c||^2 - 2 * x_r . c  (||x||^2 dropped: row-const)
// 128 rows/block; A cached in registers for entire N loop; per-lane running
// argmin; 16-lane shuffle reduction. Never materializes [65536,2048].
// Codebook tiles double-buffered via async-to-LDS on gfx1250.
// ---------------------------------------------------------------------------
__global__ __launch_bounds__(256) void vq_argmin(
    const u16t* __restrict__ Xbf,   // [NROWS, 128] bf16
    const u16t* __restrict__ CBbf,  // [2048, 128]  bf16
    const float* __restrict__ cnorm,
    int* __restrict__ idxOut)
{
  __shared__ u16t Aq[128 * 136];
  __shared__ u16t Bq[2][32 * 136];
  __shared__ float cn[KCODE];
  const int t    = threadIdx.x;
  const int lane = t & 31;
  const int wave = t >> 5;          // 8 waves, 16 rows each
  const int l16  = lane & 15;
  const int hl   = lane >> 4;
  const size_t rowBlock = (size_t)blockIdx.x * 128;

#if USE_ASYNC_LDS
  auto issueB = [&](int p, int nb) {
    #pragma unroll
    for (int i = 0; i < 2; ++i) {
      int f = t + i * 256;          // 32 codes x 16 k-segments
      int c = f >> 4, ks = f & 15;
      async_g2l_b128(CBbf + (size_t)(nb + c) * ED + ks * 8,
                     &Bq[p][c * 136 + ks * 8]);
    }
  };
  #pragma unroll
  for (int i = 0; i < 8; ++i) {
    int f = t + i * 256;            // 128 rows x 16 k-segments
    int r = f >> 4, ks = f & 15;
    async_g2l_b128(Xbf + (rowBlock + r) * (size_t)ED + ks * 8,
                   &Aq[r * 136 + ks * 8]);
  }
  issueB(0, 0);
  for (int i = t; i < KCODE; i += 256) cn[i] = cnorm[i];
  wait_async0();
  __syncthreads();
#else
  #pragma unroll
  for (int i = 0; i < 8; ++i) {
    int f = t + i * 256;
    int r = f >> 4, ks = f & 15;
    *(uint4*)&Aq[r * 136 + ks * 8] =
        *(const uint4*)(Xbf + (rowBlock + r) * (size_t)ED + ks * 8);
  }
  for (int i = t; i < KCODE; i += 256) cn[i] = cnorm[i];
  __syncthreads();
#endif

  FragBF a[4];
  #pragma unroll
  for (int s = 0; s < 4; ++s) {     // K = 128 -> 4 WMMA k-steps, held in regs
    const u16t* ap = &Aq[(wave * 16 + l16) * 136 + s * 32 + hl * 8];
    a[s].q[0] = *(const uint4*)ap;
    a[s].q[1] = *(const uint4*)(ap + 16);
  }

  float minv[8]; int mini[8];
  #pragma unroll
  for (int v = 0; v < 8; ++v) { minv[v] = 3.402823e38f; mini[v] = 0; }

  const v8f vzero = {0.f,0.f,0.f,0.f,0.f,0.f,0.f,0.f};
  int p = 0;
#if USE_ASYNC_LDS
  for (int nb = 0; nb < KCODE; nb += 32, p ^= 1) {
    if (nb + 32 < KCODE) issueB(p ^ 1, nb + 32);
#else
  for (int nb = 0; nb < KCODE; nb += 32) {
    __syncthreads();
    #pragma unroll
    for (int i = 0; i < 2; ++i) {
      int f = t + i * 256;
      int c = f >> 4, ks = f & 15;
      *(uint4*)&Bq[0][c * 136 + ks * 8] =
          *(const uint4*)(CBbf + (size_t)(nb + c) * ED + ks * 8);
    }
    __syncthreads();
#endif

    AccF acc0, acc1; acc0.v = vzero; acc1.v = vzero;
    #pragma unroll
    for (int s = 0; s < 4; ++s) {
      FragBF b0, b1;
      const u16t* bp0 = &Bq[p][(l16) * 136 + s * 32 + hl * 8];
      b0.q[0] = *(const uint4*)bp0; b0.q[1] = *(const uint4*)(bp0 + 16);
      const u16t* bp1 = &Bq[p][(16 + l16) * 136 + s * 32 + hl * 8];
      b1.q[0] = *(const uint4*)bp1; b1.q[1] = *(const uint4*)(bp1 + 16);
      acc0.v = __builtin_amdgcn_wmma_f32_16x16x32_bf16(
          false, a[s].v, false, b0.v, (short)0, acc0.v, false, false);
      acc1.v = __builtin_amdgcn_wmma_f32_16x16x32_bf16(
          false, a[s].v, false, b1.v, (short)0, acc1.v, false, false);
    }

    const int c0 = nb + l16, c1 = nb + 16 + l16;
    const float n0 = cn[c0], n1 = cn[c1];
    #pragma unroll
    for (int v = 0; v < 8; ++v) {
      float d0 = n0 - 2.0f * acc0.f[v];
      float d1 = n1 - 2.0f * acc1.f[v];
      if (d0 < minv[v]) { minv[v] = d0; mini[v] = c0; }
      if (d1 < minv[v]) { minv[v] = d1; mini[v] = c1; }
    }

#if USE_ASYNC_LDS
    wait_async0();
    __syncthreads();
#endif
  }

  #pragma unroll
  for (int v = 0; v < 8; ++v) {
    float mv = minv[v]; int mi = mini[v];
    for (int m = 1; m < 16; m <<= 1) {
      float ov = __shfl_xor(mv, m, 16);
      int   oi = __shfl_xor(mi, m, 16);
      if (ov < mv || (ov == mv && oi < mi)) { mv = ov; mi = oi; }
    }
    if (l16 == 0) idxOut[rowBlock + wave * 16 + hl * 8 + v] = mi;
  }
}

// -------------------- scatter: one-hot 1.0 + quantized gather --------------
__global__ void vq_scatter(const int* __restrict__ idx, const float* __restrict__ cb,
                           float* __restrict__ disc, float* __restrict__ quantF,
                           u16t* __restrict__ quantBF) {
  const int row = blockIdx.x, t = threadIdx.x;   // 128 threads
  const int c = idx[row];
  float v = cb[(size_t)c * ED + t];
  quantF [(size_t)row * ED + t] = v;
  quantBF[(size_t)row * ED + t] = f2bf(v);
  if (t == 0) disc[(size_t)row * KCODE + c] = 1.0f;
}

// ---------------------------------------------------------------------------
extern "C" void kernel_launch(void* const* d_in, const int* in_sizes, int n_in,
                              void* d_out, int out_size, void* d_ws, size_t ws_size,
                              hipStream_t stream) {
  (void)in_sizes; (void)n_in; (void)out_size; (void)ws_size;
  const float* cond = (const float*)d_in[0];
  const float* cb   = (const float*)d_in[1];
  const float* We1 = (const float*)d_in[2];  const float* be1 = (const float*)d_in[3];
  const float* We2 = (const float*)d_in[4];  const float* be2 = (const float*)d_in[5];
  const float* We3 = (const float*)d_in[6];  const float* be3 = (const float*)d_in[7];
  const float* We4 = (const float*)d_in[8];  const float* be4 = (const float*)d_in[9];
  const float* Wd1 = (const float*)d_in[10]; const float* bd1 = (const float*)d_in[11];
  const float* Wd2 = (const float*)d_in[12]; const float* bd2 = (const float*)d_in[13];
  const float* Wd3 = (const float*)d_in[14]; const float* bd3 = (const float*)d_in[15];
  const float* Wd4 = (const float*)d_in[16]; const float* bd4 = (const float*)d_in[17];

  // d_out layout: reconstructed | encoded | discrete | quantized
  float* out_rec  = (float*)d_out;                                 // 4096*256
  float* out_enc  = out_rec + (size_t)BDIM * PD;                   // 4096*16*128
  float* out_disc = out_enc + (size_t)NROWS * ED;                  // 65536*2048
  float* out_quant= out_disc + (size_t)NROWS * KCODE;              // 65536*128

  // workspace carve
  char* p = (char*)d_ws;
  auto carve = [&](size_t bytes) { char* r = p; p += (bytes + 255) & ~(size_t)255; return r; };
  u16t* cond_bf = (u16t*)carve((size_t)BDIM * CONDD * 2);
  u16t* actA    = (u16t*)carve((size_t)BDIM * LE * 2);
  u16t* actB    = (u16t*)carve((size_t)BDIM * LE * 2);
  u16t* qbf     = (u16t*)carve((size_t)NROWS * ED * 2);
  u16t* wte1 = (u16t*)carve((size_t)CONDD * ED * 2);
  u16t* wte2 = (u16t*)carve((size_t)ED * LE * 2);
  u16t* wte3 = (u16t*)carve((size_t)LE * LE * 2);
  u16t* wte4 = (u16t*)carve((size_t)LE * LE * 2);
  u16t* wtd1 = (u16t*)carve((size_t)LE * LE * 2);
  u16t* wtd2 = (u16t*)carve((size_t)LE * LE * 2);
  u16t* wtd3 = (u16t*)carve((size_t)LE * ED * 2);
  u16t* wtd4 = (u16t*)carve((size_t)ED * PD * 2);
  u16t* cbbf = (u16t*)carve((size_t)KCODE * ED * 2);
  float* cnorm = (float*)carve((size_t)KCODE * 4);
  int*   idx   = (int*)carve((size_t)NROWS * 4);

  // ---- precision conversions / transposes ----
  {
    int n = BDIM * CONDD;
    cvt_f32_bf16<<<(n + 255) / 256, 256, 0, stream>>>(cond, cond_bf, n);
  }
  dim3 tb(32, 8);
  wxpose_bf16<<<dim3(ED / 32,   CONDD / 32), tb, 0, stream>>>(We1, wte1, CONDD, ED);
  wxpose_bf16<<<dim3(LE / 32,   ED / 32),    tb, 0, stream>>>(We2, wte2, ED,    LE);
  wxpose_bf16<<<dim3(LE / 32,   LE / 32),    tb, 0, stream>>>(We3, wte3, LE,    LE);
  wxpose_bf16<<<dim3(LE / 32,   LE / 32),    tb, 0, stream>>>(We4, wte4, LE,    LE);
  wxpose_bf16<<<dim3(LE / 32,   LE / 32),    tb, 0, stream>>>(Wd1, wtd1, LE,    LE);
  wxpose_bf16<<<dim3(LE / 32,   LE / 32),    tb, 0, stream>>>(Wd2, wtd2, LE,    LE);
  wxpose_bf16<<<dim3(ED / 32,   LE / 32),    tb, 0, stream>>>(Wd3, wtd3, LE,    ED);
  wxpose_bf16<<<dim3(PD / 32,   ED / 32),    tb, 0, stream>>>(Wd4, wtd4, ED,    PD);
  prep_codebook<<<KCODE, 128, 0, stream>>>(cb, cbbf, cnorm);

  // ---- encoder ----
  gemm_bf16_lrelu<<<dim3(ED / 128, BDIM / 128), 256, 0, stream>>>(cond_bf, wte1, be1, actA, nullptr, ED, CONDD);
  gemm_bf16_lrelu<<<dim3(LE / 128, BDIM / 128), 256, 0, stream>>>(actA, wte2, be2, actB, nullptr, LE, ED);
  gemm_bf16_lrelu<<<dim3(LE / 128, BDIM / 128), 256, 0, stream>>>(actB, wte3, be3, actA, nullptr, LE, LE);
  gemm_bf16_lrelu<<<dim3(LE / 128, BDIM / 128), 256, 0, stream>>>(actA, wte4, be4, actB, out_enc, LE, LE);

  // ---- VQ ----
  zero_f32v4<<<2048, 256, 0, stream>>>((float4*)out_disc, (size_t)NROWS * KCODE / 4);
  vq_argmin<<<NROWS / 128, 256, 0, stream>>>(actB, cbbf, cnorm, idx);
  vq_scatter<<<NROWS, 128, 0, stream>>>(idx, cb, out_disc, out_quant, qbf);

  // ---- decoder ----
  gemm_bf16_lrelu<<<dim3(LE / 128, BDIM / 128), 256, 0, stream>>>(qbf,  wtd1, bd1, actA, nullptr, LE, LE);
  gemm_bf16_lrelu<<<dim3(LE / 128, BDIM / 128), 256, 0, stream>>>(actA, wtd2, bd2, actB, nullptr, LE, LE);
  gemm_bf16_lrelu<<<dim3(ED / 128, BDIM / 128), 256, 0, stream>>>(actB, wtd3, bd3, actA, nullptr, ED, LE);
  gemm_bf16_lrelu<<<dim3(PD / 128, BDIM / 128), 256, 0, stream>>>(actA, wtd4, bd4, actB, out_rec, PD, ED);
}
